// GCN_30846455120743
// MI455X (gfx1250) — matrix operand order repaired and checked by reading, not compiled, not run on hardware
//
#include <hip/hip_runtime.h>
#include <hip/hip_bf16.h>

typedef _Float16 f16;
typedef __attribute__((ext_vector_type(16))) _Float16 v16h;
typedef __attribute__((ext_vector_type(2)))  _Float16 v2h;
typedef __attribute__((ext_vector_type(8)))  float    v8f;

// A-fragment layout: within each 32-wide K block, features stay in natural order
// [K0..15 | K16..31] so that lane-half hl loads its entire 16-element WMMA A
// fragment as ONE contiguous 32-byte load (contiguous k order, matching B).
__device__ __forceinline__ int aswz(int f) {
    return f;
}

// ---------------- small utility kernels ----------------

__global__ void k_fill_f32(float* p, float v, int n) {
    int t = blockIdx.x * blockDim.x + threadIdx.x;
    if (t < n) p[t] = v;
}

__global__ void k_zero_f32(float* p, int n) {
    int t = blockIdx.x * blockDim.x + threadIdx.x;
    if (t < n) p[t] = 0.0f;
}

__global__ void k_deg_accum(const int* __restrict__ dst, float* __restrict__ deg, int E) {
    int e = blockIdx.x * blockDim.x + threadIdx.x;
    if (e < E) atomicAdd(&deg[dst[e]], 1.0f);
}

__global__ void k_dinv(const float* __restrict__ deg, float* __restrict__ dinv, int N) {
    int i = blockIdx.x * blockDim.x + threadIdx.x;
    if (i < N) {
        float d = deg[i];                       // always >= 1 (self loop)
        dinv[i] = __frsqrt_rn(d < 1.0f ? 1.0f : d);
    }
}

__global__ void k_norm(const int* __restrict__ src, const int* __restrict__ dst,
                       const float* __restrict__ dinv, float* __restrict__ nrm, int E) {
    int e = blockIdx.x * blockDim.x + threadIdx.x;
    if (e < E) nrm[e] = dinv[src[e]] * dinv[dst[e]];
}

// x (f32, row-major [N,128]) -> Xh (f16, A-swizzled rows)
__global__ void k_cvt_f16_swz(const float* __restrict__ in, f16* __restrict__ out, int n) {
    int t = blockIdx.x * blockDim.x + threadIdx.x;
    if (t >= n) return;
    int f = t & 127;
    out[(t & ~127) | aswz(f)] = (f16)in[t];
}

// W [fi, fo] row-major (f32) -> Wt [fo, fi] row-major (f16) (B fragment = contiguous)
__global__ void k_wt_cvt(const float* __restrict__ W, f16* __restrict__ Wt, int fi, int fo) {
    int t = blockIdx.x * blockDim.x + threadIdx.x;
    if (t >= fi * fo) return;
    int i = t / fo, o = t % fo;
    Wt[o * fi + i] = (f16)W[t];
}

// ---------------- WMMA GEMM: Hh[N,fo] = Xh[N,FI] @ W ----------------
// Compile-time FI (K depth) and NT (16-col tiles per wave): K loop fully
// unrolls into straight-line v_wmma ops, accumulators pinned in registers.
template<int FI, int NT>
__global__ void __launch_bounds__(128)
k_gemm_wmma(const f16* __restrict__ Xh, const f16* __restrict__ Wt,
            f16* __restrict__ Hh, int N, int fo) {
    const int lane  = threadIdx.x & 31;
    const int wave  = threadIdx.x >> 5;
    const int mBase = (blockIdx.x * 4 + wave) * 16;
    if (mBase >= N) return;                      // wave-uniform: EXEC stays all-ones
    const int nBase = blockIdx.y * (NT * 16);
    const int row   = lane & 15;
    const int hl    = lane >> 4;                 // lane half: 0 or 1

    const f16* arow = Xh + (size_t)(mBase + row) * FI;   // swizzled rows
    const f16* bt[NT];
#pragma unroll
    for (int nt = 0; nt < NT; ++nt)
        bt[nt] = Wt + (size_t)(nBase + nt * 16 + row) * FI;

    v8f c[NT];
#pragma unroll
    for (int nt = 0; nt < NT; ++nt) c[nt] = (v8f){};

#pragma unroll
    for (int kt = 0; kt < FI; kt += 32) {
        v16h a = *(const v16h*)(arow + kt + hl * 16);        // one 32B load (swizzled)
#pragma unroll
        for (int nt = 0; nt < NT; ++nt) {
            v16h b = *(const v16h*)(bt[nt] + kt + hl * 16);  // one 32B load (transposed W)
            c[nt] = __builtin_amdgcn_wmma_f32_16x16x32_f16(
                false, a, false, b, (short)0, c[nt], false, false);
        }
    }

    // D layout: VGPR r holds M = mBase + hl*8 + r, N = nBase + nt*16 + (lane&15)
#pragma unroll
    for (int nt = 0; nt < NT; ++nt) {
        int ncol = nBase + nt * 16 + row;
        if (ncol < fo) {
#pragma unroll
            for (int r = 0; r < 8; ++r) {
                int m = mBase + hl * 8 + r;
                Hh[(size_t)m * fo + ncol] = (f16)c[nt][r];
            }
        }
    }
}

// ---------------- edge gather / scale / scatter-add ----------------
// One wave per edge; lanes stride features in half2 granularity.
// f16 gather halves the dominant (L2-resident) traffic; f32 atomics keep accuracy.
__global__ void k_edge_scatter(const f16* __restrict__ Hh, const int* __restrict__ src,
                               const int* __restrict__ dst, const float* __restrict__ nrm,
                               float* __restrict__ acc, int E, int fo) {
    long long t = (long long)blockIdx.x * blockDim.x + threadIdx.x;
    int e    = (int)(t >> 5);
    int lane = (int)(t & 31);
    if (e >= E) return;
    int s = src[e], d = dst[e];
    float w = nrm[e];
    const f16* hrow = Hh  + (size_t)s * fo;
    float*     orow = acc + (size_t)d * fo;
    for (int f = lane * 2; f < fo; f += 64) {       // fo is even (128 or 40)
        v2h h = *(const v2h*)(hrow + f);
        atomicAdd(orow + f,     (float)h[0] * w);
        atomicAdd(orow + f + 1, (float)h[1] * w);
    }
}

// ---------------- finalize: self-loop + bias + ReLU -> swizzled f16 ----------------
__global__ void k_finalize_relu(const float* __restrict__ acc, const f16* __restrict__ Hh,
                                const float* __restrict__ dinv, const float* __restrict__ bias,
                                f16* __restrict__ Xout, int N, int fo) {
    int t = blockIdx.x * blockDim.x + threadIdx.x;
    if (t >= N * fo) return;
    int i = t / fo, f = t % fo;
    float di = dinv[i];
    float v = acc[t] + (float)Hh[t] * di * di + bias[f];
    Xout[i * fo + aswz(f)] = (f16)(v > 0.0f ? v : 0.0f);    // fo==128 here
}

__global__ void k_finalize_last(float* __restrict__ acc, const f16* __restrict__ Hh,
                                const float* __restrict__ dinv, const float* __restrict__ bias,
                                int N, int fo) {
    int t = blockIdx.x * blockDim.x + threadIdx.x;
    if (t >= N * fo) return;
    int i = t / fo, f = t % fo;
    float di = dinv[i];
    acc[t] = acc[t] + (float)Hh[t] * di * di + bias[f];     // in-place, elementwise-safe
}

// ---------------- log-softmax: one wave32 per node ----------------
__global__ void k_log_softmax(const float* __restrict__ logits, float* __restrict__ out,
                              int N, int C) {
    int node = blockIdx.x * (blockDim.x >> 5) + (threadIdx.x >> 5);
    int lane = threadIdx.x & 31;
    if (node >= N) return;
    const float* row = logits + (size_t)node * C;
    const float NEG = -__builtin_inff();
    float v0 = (lane      < C) ? row[lane]      : NEG;
    float v1 = (lane + 32 < C) ? row[lane + 32] : NEG;
    float m = fmaxf(v0, v1);
#pragma unroll
    for (int off = 16; off > 0; off >>= 1) m = fmaxf(m, __shfl_xor(m, off, 32));
    float s = ((lane < C) ? expf(v0 - m) : 0.0f) + ((lane + 32 < C) ? expf(v1 - m) : 0.0f);
#pragma unroll
    for (int off = 16; off > 0; off >>= 1) s += __shfl_xor(s, off, 32);
    float ls = logf(s);
    if (lane      < C) out[(size_t)node * C + lane]      = v0 - m - ls;
    if (lane + 32 < C) out[(size_t)node * C + lane + 32] = v1 - m - ls;
}

// ---------------- host-side orchestration ----------------

extern "C" void kernel_launch(void* const* d_in, const int* in_sizes, int n_in,
                              void* d_out, int out_size, void* d_ws, size_t ws_size,
                              hipStream_t stream) {
    const int FIN = 128, HID = 128, C = 40;
    const float* x  = (const float*)d_in[0];
    const int*   ei = (const int*)d_in[1];
    const int N = in_sizes[0] / FIN;
    const int E = in_sizes[1] / 2;
    const int* src = ei;
    const int* dst = ei + E;

    const float *W[6], *B[6];
    for (int l = 0; l < 6; ++l) { W[l] = (const float*)d_in[2 + 2 * l];
                                  B[l] = (const float*)d_in[3 + 2 * l]; }

    // Workspace carve (all buffers L2-resident: ~55 MB total)
    char* p = (char*)d_ws;
    auto carve = [&](size_t bytes) { char* r = p; p += (bytes + 255) & ~(size_t)255; return r; };
    f16*   Xh   = (f16*)  carve((size_t)N * HID * sizeof(f16));
    f16*   Hh   = (f16*)  carve((size_t)N * HID * sizeof(f16));
    float* acc  = (float*)carve((size_t)N * HID * sizeof(float));
    f16*   Wt   = (f16*)  carve((size_t)HID * HID * sizeof(f16));
    float* deg  = (float*)carve((size_t)N * sizeof(float));
    float* dinv = (float*)carve((size_t)N * sizeof(float));
    float* nrm  = (float*)carve((size_t)E * sizeof(float));
    (void)ws_size; (void)n_in; (void)out_size;

    const int TB = 256;
    auto blk = [&](long long n) { return (int)((n + TB - 1) / TB); };

    // GCN symmetric norm (computed once, reused by all 6 layers)
    k_fill_f32 <<<blk(N), TB, 0, stream>>>(deg, 1.0f, N);         // self loops
    k_deg_accum<<<blk(E), TB, 0, stream>>>(dst, deg, E);
    k_dinv     <<<blk(N), TB, 0, stream>>>(deg, dinv, N);
    k_norm     <<<blk(E), TB, 0, stream>>>(src, dst, dinv, nrm, E);

    // x -> f16 (A-swizzled) once
    k_cvt_f16_swz<<<blk((long long)N * FIN), TB, 0, stream>>>(x, Xh, N * FIN);

    for (int l = 0; l < 6; ++l) {
        const int fo = (l == 5) ? C : HID;

        k_wt_cvt<<<blk(HID * fo), TB, 0, stream>>>(W[l], Wt, HID, fo);

        const int mBlocks = (N / 16 + 3) / 4;
        if (l < 5) {
            dim3 gg(mBlocks, (HID + 63) / 64);                     // 2 n-slabs of 64
            k_gemm_wmma<128, 4><<<gg, 128, 0, stream>>>(Xh, Wt, Hh, N, fo);
        } else {
            dim3 gg(mBlocks, 1);                                   // 48 cols, guard to 40
            k_gemm_wmma<128, 3><<<gg, 128, 0, stream>>>(Xh, Wt, Hh, N, fo);
        }

        k_zero_f32<<<blk((long long)N * fo), TB, 0, stream>>>(acc, N * fo);

        long long et = (long long)E * 32;
        k_edge_scatter<<<blk(et), TB, 0, stream>>>(Hh, src, dst, nrm, acc, E, fo);

        if (l < 5)
            k_finalize_relu<<<blk((long long)N * fo), TB, 0, stream>>>(acc, Hh, dinv, B[l], Xh, N, fo);
        else
            k_finalize_last<<<blk((long long)N * fo), TB, 0, stream>>>(acc, Hh, dinv, B[l], N, fo);
    }

    k_log_softmax<<<(N + 7) / 8, 256, 0, stream>>>(acc, (float*)d_out, N, C);
}
